// BAD_GraphColoringTransformer_17016660427428
// MI455X (gfx1250) — compile-verified, matrix operand order checked by reading, not run on hardware
//
#include <hip/hip_runtime.h>
#include <hip/hip_bf16.h>
#include <math.h>

// ---------------------------------------------------------------------------
// MI455X (gfx1250) GCN/TransformerConv pipeline.
//  - Dense GEMMs: v_wmma_f32_16x16x32_bf16; A tile via vectorized global->LDS,
//    W tile via Tensor Data Mover (tensor_load_to_lds + s_wait_tensorcnt).
//  - Fragments loaded as ds_load_b128 (row-major A / transposed-B LDS layout).
//  - Edge phases: scalar + global float atomics (bandwidth bound).
// ---------------------------------------------------------------------------

typedef __attribute__((ext_vector_type(16))) __bf16 v16bf;
typedef __attribute__((ext_vector_type(8)))  float  v8f;
typedef __attribute__((ext_vector_type(8)))  unsigned short ushort8;
typedef __attribute__((ext_vector_type(4)))  unsigned int u32x4;
typedef __attribute__((ext_vector_type(8)))  int i32x8;
typedef __attribute__((ext_vector_type(4)))  int i32x4;

__device__ __forceinline__ unsigned short f32_to_bf16(float f) {
  union { float f; unsigned u; } c; c.f = f;
  unsigned r = c.u + 0x7FFFu + ((c.u >> 16) & 1u);   // round-to-nearest-even
  return (unsigned short)(r >> 16);
}

// Issue a 2D TDM tile load (f32 elements) into LDS. D# packing per ISA 8.3/8.4.
// 6-arg builtin form (clang-23 / therock-10.0 headers).
__device__ __forceinline__ void tdm_load_tile_f32(
    const float* gaddr, unsigned ldsOff,
    unsigned remX, unsigned remY, unsigned tileX, unsigned tileY,
    unsigned strideX)
{
  unsigned long long ga = (unsigned long long)(uintptr_t)gaddr;
  u32x4 g0;
  g0[0] = 1u;                                        // count=1 (valid user D#)
  g0[1] = ldsOff;                                    // lds_addr
  g0[2] = (unsigned)(ga & 0xFFFFFFFFu);              // global_addr[31:0]
  g0[3] = (unsigned)((ga >> 32) & 0x1FFFFFFu)        // global_addr[56:32]
        | 0x80000000u;                               // type=2 ("image")
  i32x8 g1;
  g1[0] = (int)(2u << 16);                           // data_size=2 -> 4 bytes
  g1[1] = (int)((remX & 0xFFFFu) << 16);             // tensor_dim0[15:0]
  g1[2] = (int)((remX >> 16) | ((remY & 0xFFFFu) << 16)); // dim0 hi | dim1 lo
  g1[3] = (int)((remY >> 16) | (tileX << 16));       // dim1 hi | tile_dim0
  g1[4] = (int)tileY;                                // tile_dim1 (tile_dim2=0)
  g1[5] = (int)strideX;                              // tensor_dim0_stride lo
  g1[6] = 0;                                         // stride hi / dim1_stride lo
  g1[7] = 0;
  i32x4 z4 = {0, 0, 0, 0};
  i32x8 z8 = {0, 0, 0, 0, 0, 0, 0, 0};
  __builtin_amdgcn_tensor_load_to_lds(g0, g1, z4, z4, z8, 0);
}

// ------------------------- WMMA GEMM: C = A @ W (+bias) --------------------
// A: [M x K] f32 row-major, W: [K x N] f32 row-major, C: [M x N] f32.
// Block = 128 threads = 4 waves; each wave computes one 16x16 C tile;
// block covers 64 rows x 16 cols. K loop in steps of 32 (zero padded).
__global__ __launch_bounds__(128)
void gemm_bf16_wmma(const float* __restrict__ A, const float* __restrict__ W,
                    const float* __restrict__ bias, float* __restrict__ C,
                    int M, int K, int N, int useBias) {
  __shared__ unsigned short sA[64][32];   // A tile, row-major, 16B-aligned rows
  __shared__ unsigned short sBT[16][32];  // W tile transposed: [n][k]
  __shared__ float sW32[32][16];          // f32 staging filled by the TDM

  const int tid  = threadIdx.x;
  const int lane = tid & 31;
  const int wave = tid >> 5;
  const int rowBase = blockIdx.x * 64;
  const int colBase = blockIdx.y * 16;

  v8f acc = {0.f, 0.f, 0.f, 0.f, 0.f, 0.f, 0.f, 0.f};
  const int nK = (K + 31) >> 5;

  for (int kt = 0; kt < nK; ++kt) {
    const int k0 = kt << 5;

    // ---- W tile via Tensor Data Mover (one issue per block, wave 0) ----
    if (tid == 0) {
      tdm_load_tile_f32(W + (size_t)k0 * N + colBase,
                        (unsigned)(uintptr_t)&sW32[0][0],
                        (unsigned)(N - colBase), (unsigned)(K - k0),
                        16u, 32u, (unsigned)N);
    }

    // ---- A tile: global -> LDS with f32->bf16 conversion ----
    if (rowBase + 64 <= M && k0 + 32 <= K) {
      // fast path: fully in-bounds, 128-bit loads
      for (int i = tid; i < 512; i += 128) {
        int r = i >> 3, c = (i & 7) << 2;
        float4 v = *(const float4*)&A[(size_t)(rowBase + r) * K + k0 + c];
        sA[r][c + 0] = f32_to_bf16(v.x);
        sA[r][c + 1] = f32_to_bf16(v.y);
        sA[r][c + 2] = f32_to_bf16(v.z);
        sA[r][c + 3] = f32_to_bf16(v.w);
      }
    } else {
      // slow path: branchless clamped loads + select
      for (int i = tid; i < 2048; i += 128) {
        int r = i >> 5, kk = i & 31;
        int gr = rowBase + r, gk = k0 + kk;
        int ok = (gr < M) && (gk < K);
        int cr = gr < M ? gr : (M - 1);
        int ck = gk < K ? gk : (K - 1);
        float v = A[(size_t)cr * K + ck];
        sA[r][kk] = f32_to_bf16(ok ? v : 0.f);
      }
    }
    // prefetch next A K-slab into L2 (global_prefetch_b8)
    if (kt + 1 < nK) {
      int pr = rowBase + (tid & 63);
      if (pr < M) __builtin_prefetch(&A[(size_t)pr * K + k0 + 32], 0, 1);
    }

    __builtin_amdgcn_s_wait_tensorcnt(0);   // TDM done (no-op for waves 1..3)
    __syncthreads();                        // sA + sW32 visible to all

    // ---- convert + transpose W tile: sW32[k][n] -> sBT[n][k] bf16 ----
    for (int i = tid; i < 512; i += 128) {
      int kk = i >> 4, n = i & 15;
      float v = (k0 + kk < K && colBase + n < N) ? sW32[kk][n] : 0.f;
      sBT[n][kk] = f32_to_bf16(v);
    }
    __syncthreads();

    // ---- fragments: two ds_load_b128 each (ISA 7.12.2 row-major layout) ----
    const int half = lane >> 4;
    const int mr   = lane & 15;
    const int arow = wave * 16 + mr;
    union { ushort8 h[2]; v16bf v; } fa, fb;
    fa.h[0] = *(const ushort8*)&sA[arow][half * 8];        // K = h*8 .. h*8+7
    fa.h[1] = *(const ushort8*)&sA[arow][16 + half * 8];   // K = 16+h*8 ..
    fb.h[0] = *(const ushort8*)&sBT[mr][half * 8];
    fb.h[1] = *(const ushort8*)&sBT[mr][16 + half * 8];

    acc = __builtin_amdgcn_wmma_f32_16x16x32_bf16(
        /*neg_a=*/false, fa.v, /*neg_b=*/false, fb.v,
        /*c_mod=*/(short)0, acc, /*reuse_a=*/false, /*reuse_b=*/false);
    __syncthreads();
  }

  // Store C per 32-bit C/D layout: VGPR r -> M = r + 8*half, N = lane&15.
  const int half = lane >> 4;
  const int mr   = lane & 15;
#pragma unroll
  for (int r = 0; r < 8; ++r) {
    int gm = rowBase + wave * 16 + half * 8 + r;
    int gn = colBase + mr;
    if (gm < M && gn < N) {
      float v = acc[r];
      if (useBias) v += bias[gn];
      C[(size_t)gm * N + gn] = v;
    }
  }
}

// ------------------------------ utility kernels ----------------------------
__global__ void fill_kernel(float* p, long long n, float v) {
  long long t = (long long)blockIdx.x * blockDim.x + threadIdx.x;
  if (t < n) p[t] = v;
}

__global__ void relu_kernel(float* x, long long n) {
  long long t = (long long)blockIdx.x * blockDim.x + threadIdx.x;
  if (t < n) x[t] = fmaxf(x[t], 0.f);
}

__global__ void deg_kernel(const int* __restrict__ dst, float* deg, int E) {
  int e = blockIdx.x * blockDim.x + threadIdx.x;
  if (e < E) atomicAdd(&deg[dst[e]], 1.0f);
}

__global__ void dinv_kernel(const float* __restrict__ deg, float* dinv, int Nn) {
  int n = blockIdx.x * blockDim.x + threadIdx.x;
  if (n < Nn) dinv[n] = rsqrtf(deg[n] + 1.0f);
}

// scatter-add of normalized messages; grid.y = feature index
__global__ void gcn_scatter_kernel(const float* __restrict__ h,
                                   const int* __restrict__ src,
                                   const int* __restrict__ dst,
                                   const float* __restrict__ dinv,
                                   float* out, int E, int D) {
  int e = blockIdx.x * blockDim.x + threadIdx.x;
  int f = blockIdx.y;
  if (e >= E) return;
  int s = src[e], d0 = dst[e];
  float c = dinv[s] * dinv[d0];
  atomicAdd(&out[(size_t)d0 * D + f], h[(size_t)s * D + f] * c);
}

__global__ void gcn_finish_kernel(const float* __restrict__ h,
                                  const float* __restrict__ dinv,
                                  const float* __restrict__ b,
                                  float* out, int Nn, int D, int doRelu) {
  long long t = (long long)blockIdx.x * blockDim.x + threadIdx.x;
  if (t >= (long long)Nn * D) return;
  int n = (int)(t / D), f = (int)(t % D);
  float di = dinv[n];
  float o = out[t] + h[t] * di * di + b[f];
  out[t] = doRelu ? fmaxf(o, 0.f) : o;
}

__global__ void row_softmax_kernel(float* x, int Nn, int D) {
  int r = blockIdx.x * blockDim.x + threadIdx.x;
  if (r >= Nn) return;
  float* xr = x + (size_t)r * D;
  float mx = xr[0];
  for (int f = 1; f < D; ++f) mx = fmaxf(mx, xr[f]);
  float s = 0.f;
  for (int f = 0; f < D; ++f) { float e = __expf(xr[f] - mx); xr[f] = e; s += e; }
  float inv = 1.f / s;
  for (int f = 0; f < D; ++f) xr[f] *= inv;
}

__device__ __forceinline__ void atomicMaxF(float* addr, float val) {
  if (val >= 0.f) atomicMax((int*)addr, __float_as_int(val));
  else            atomicMin((unsigned int*)addr, __float_as_uint(val));
}

__global__ void tr_score_kernel(const float* __restrict__ q,
                                const float* __restrict__ k,
                                const int* __restrict__ src,
                                const int* __restrict__ dst,
                                float* esc, float* nmax, int E, int D, float scale) {
  int e = blockIdx.x * blockDim.x + threadIdx.x;
  if (e >= E) return;
  const float* qd = q + (size_t)dst[e] * D;
  const float* ks = k + (size_t)src[e] * D;
  float s = 0.f;
  if ((D & 3) == 0) {
    for (int f = 0; f < D; f += 4) {
      float4 a = *(const float4*)&qd[f];
      float4 b = *(const float4*)&ks[f];
      s += a.x * b.x + a.y * b.y + a.z * b.z + a.w * b.w;
    }
  } else {
    for (int f = 0; f < D; ++f) s += qd[f] * ks[f];
  }
  s *= scale;
  esc[e] = s;
  atomicMaxF(&nmax[dst[e]], s);
}

__global__ void tr_exp_kernel(const float* __restrict__ nmax,
                              const int* __restrict__ dst,
                              float* esc, float* ndenom, int E) {
  int e = blockIdx.x * blockDim.x + threadIdx.x;
  if (e >= E) return;
  float ex = __expf(esc[e] - nmax[dst[e]]);
  esc[e] = ex;
  atomicAdd(&ndenom[dst[e]], ex);
}

// alpha-weighted scatter of v[src]; grid.y = feature index
__global__ void tr_agg_kernel(const float* __restrict__ vbuf,
                              const float* __restrict__ esc,
                              const float* __restrict__ ndenom,
                              const int* __restrict__ src,
                              const int* __restrict__ dst,
                              float* out, int E, int D) {
  int e = blockIdx.x * blockDim.x + threadIdx.x;
  int f = blockIdx.y;
  if (e >= E) return;
  int d0 = dst[e];
  float alpha = esc[e] / (ndenom[d0] + 1e-16f);
  atomicAdd(&out[(size_t)d0 * D + f], alpha * vbuf[(size_t)src[e] * D + f]);
}

// x5 = softmax(h[:, :5]); hout = concat(x5, h[:, 5:]) for width-10 h
__global__ void split_softmax5_kernel(const float* __restrict__ hin,
                                      float* x5, float* hout, int Nn) {
  int r = blockIdx.x * blockDim.x + threadIdx.x;
  if (r >= Nn) return;
  const float* hi = hin + (size_t)r * 10;
  float mx = hi[0];
  for (int f = 1; f < 5; ++f) mx = fmaxf(mx, hi[f]);
  float e[5], s = 0.f;
  for (int f = 0; f < 5; ++f) { e[f] = __expf(hi[f] - mx); s += e[f]; }
  float inv = 1.f / s;
  for (int f = 0; f < 5; ++f) {
    float v = e[f] * inv;
    x5[(size_t)r * 5 + f] = v;
    hout[(size_t)r * 10 + f] = v;
  }
  for (int f = 5; f < 10; ++f) hout[(size_t)r * 10 + f] = hi[f];
}

// ------------------------------- host driver -------------------------------
static inline int cdiv(long long a, long long b) { return (int)((a + b - 1) / b); }

extern "C" void kernel_launch(void* const* d_in, const int* in_sizes, int n_in,
                              void* d_out, int out_size, void* d_ws, size_t ws_size,
                              hipStream_t stream) {
  const float* x  = (const float*)d_in[0];
  const int*   ei = (const int*)d_in[1];
  const int Nn = in_sizes[0] / 10;
  const int E  = in_sizes[1] / 2;
  const int H  = 128;
  const int* src = ei;
  const int* dst = ei + E;

  float* ws = (float*)d_ws;
  const size_t NH = (size_t)Nn * H;
  float* act0   = ws;
  float* act1   = ws + 1 * NH;
  float* b2     = ws + 2 * NH;   // gcn h / q
  float* b3     = ws + 3 * NH;   // k
  float* b4     = ws + 4 * NH;   // v
  float* dinv   = ws + 5 * NH;
  float* nmax   = dinv + Nn;     // also deg temp
  float* ndenom = nmax + Nn;
  float* esc    = ndenom + Nn;   // E floats

  // ---- degrees (shared by all GCN layers) ----
  fill_kernel<<<cdiv(Nn, 256), 256, 0, stream>>>(nmax, Nn, 0.f);
  deg_kernel<<<cdiv(E, 256), 256, 0, stream>>>(dst, nmax, E);
  dinv_kernel<<<cdiv(Nn, 256), 256, 0, stream>>>(nmax, dinv, Nn);

  auto gemm = [&](const float* A, const float* W, const float* b, float* C,
                  int M, int K, int Ncol, int useBias) {
    dim3 g((unsigned)cdiv(M, 64), (unsigned)cdiv(Ncol, 16));
    gemm_bf16_wmma<<<g, 128, 0, stream>>>(A, W, b, C, M, K, Ncol, useBias);
  };

  auto gcn = [&](int li, const float* in, int din, int dout, float* out,
                 int doRelu, int doSm) {
    const float* Wp = (const float*)d_in[2 + 2 * li];
    const float* bp = (const float*)d_in[3 + 2 * li];
    gemm(in, Wp, nullptr, b2, Nn, din, dout, 0);
    long long tot = (long long)Nn * dout;
    fill_kernel<<<cdiv(tot, 256), 256, 0, stream>>>(out, tot, 0.f);
    dim3 gs((unsigned)cdiv(E, 256), (unsigned)dout);
    gcn_scatter_kernel<<<gs, 256, 0, stream>>>(b2, src, dst, dinv, out, E, dout);
    gcn_finish_kernel<<<cdiv(tot, 256), 256, 0, stream>>>(b2, dinv, bp, out, Nn, dout, doRelu);
    if (doSm) row_softmax_kernel<<<cdiv(Nn, 256), 256, 0, stream>>>(out, Nn, dout);
  };

  auto tconv = [&](int lj, const float* in, int din, int dout, float* out, int doSm) {
    int base = 20 + 8 * lj;
    const float* Wq = (const float*)d_in[base + 0]; const float* bq = (const float*)d_in[base + 1];
    const float* Wk = (const float*)d_in[base + 2]; const float* bk = (const float*)d_in[base + 3];
    const float* Wv = (const float*)d_in[base + 4]; const float* bv = (const float*)d_in[base + 5];
    const float* Ws = (const float*)d_in[base + 6]; const float* bs = (const float*)d_in[base + 7];
    gemm(in, Wq, bq, b2, Nn, din, dout, 1);
    gemm(in, Wk, bk, b3, Nn, din, dout, 1);
    gemm(in, Wv, bv, b4, Nn, din, dout, 1);
    gemm(in, Ws, bs, out, Nn, din, dout, 1);   // skip connection seeds out
    fill_kernel<<<cdiv(Nn, 256), 256, 0, stream>>>(nmax, Nn, -INFINITY);
    fill_kernel<<<cdiv(Nn, 256), 256, 0, stream>>>(ndenom, Nn, 0.f);
    float scale = 1.0f / sqrtf((float)dout);
    tr_score_kernel<<<cdiv(E, 256), 256, 0, stream>>>(b2, b3, src, dst, esc, nmax, E, dout, scale);
    tr_exp_kernel<<<cdiv(E, 256), 256, 0, stream>>>(nmax, dst, esc, ndenom, E);
    dim3 ga((unsigned)cdiv(E, 256), (unsigned)dout);
    tr_agg_kernel<<<ga, 256, 0, stream>>>(b4, esc, ndenom, src, dst, out, E, dout);
    long long tot = (long long)Nn * dout;
    relu_kernel<<<cdiv(tot, 256), 256, 0, stream>>>(out, tot);
    if (doSm) row_softmax_kernel<<<cdiv(Nn, 256), 256, 0, stream>>>(out, Nn, dout);
  };

  // ---------------- pipeline ----------------
  gcn(0, x,    10, 128, act0, 1, 0);
  gcn(1, act0, 128, 128, act1, 1, 0);
  gcn(2, act1, 128, 10,  act0, 0, 1);

  tconv(0, act0, 10, 128, act1, 0);
  tconv(1, act1, 128, 128, act0, 0);
  tconv(2, act0, 128, 10,  act1, 1);
  tconv(3, act1, 10, 128, act0, 0);
  tconv(4, act0, 128, 128, act1, 0);
  tconv(5, act1, 128, 10,  act0, 1);

  gcn(3, act0, 10, 128, act1, 1, 0);
  gcn(4, act1, 128, 128, act0, 1, 0);
  gcn(5, act0, 128, 10,  act1, 0, 1);

  float* x5_out = (float*)d_out;                    // [Nn x 5]
  float* h_out  = (float*)d_out + (size_t)Nn * 5;   // [Nn x 10]
  split_softmax5_kernel<<<cdiv(Nn, 256), 256, 0, stream>>>(act1, x5_out, act0, Nn);

  gcn(6, act0, 10, 128, act1, 1, 0);
  gcn(7, act1, 128, 128, act0, 1, 0);
  gcn(8, act0, 128, 10,  h_out, 0, 1);
}
